// RNNPB_65824668778548
// MI455X (gfx1250) — matrix operand order repaired and checked by reading, not compiled, not run on hardware
//
#include <hip/hip_runtime.h>

// RNNPB persistent-kernel for MI455X (gfx1250, wave32).
// One workgroup (16 waves) on one WGP carries the whole T=8192 recurrence.
// Weights live in VGPRs as WMMA B-fragments; ctx/h ping-pong through LDS.

#define B_N      30     // batch (padded to 32 = 2 M-tiles)
#define D_DIM    6
#define H_DIM    240    // 15 n-tiles for GEMM1
#define C_DIM    180
#define T_LEN    8192
#define K1       192    // ctx(180) + x(6) + pad -> 6 k-chunks of 32
#define K2       256    // h(240) + pad          -> 8 k-chunks of 32
#define XCH      64     // timesteps of input staged per LDS refill
#define NTHREADS 512    // 16 wave32 waves

typedef __attribute__((ext_vector_type(16))) _Float16 v16h;
typedef __attribute__((ext_vector_type(8)))  _Float16 v8h;
typedef __attribute__((ext_vector_type(8)))  float    v8f;

// Load one 16x32 f16 A-fragment from a row-major LDS matrix.
// Layout: lane L (<16): M=L, K = 0..15 of the chunk;
// lane L+16: M=L, K = 16..31.  -> two ds_load_b128 per fragment.
__device__ __forceinline__ v16h load_a_frag(const _Float16* base, int rowstride,
                                            int row0, int kc, int lane) {
  int row   = row0 + (lane & 15);
  int kbase = kc * 32 + (lane >> 4) * 16;
  const _Float16* p = base + row * rowstride + kbase;
  union { v16h v; v8h h[2]; } u;
  u.h[0] = *(const v8h*)(p);        // K chunk lo
  u.h[1] = *(const v8h*)(p + 8);    // K chunk hi (+8 halves)
  return u.v;
}

__device__ __forceinline__ v8f wmma16(v16h a, v16h b, v8f c) {
  // v_wmma_f32_16x16x32_f16  (8-arg form: neg_a, A, neg_b, B, c_mod, C, reuse_a, reuse_b)
  return __builtin_amdgcn_wmma_f32_16x16x32_f16(false, a, false, b, (short)0, c,
                                                false, false);
}

__global__ __launch_bounds__(NTHREADS)
void rnnpb_persistent(const float* __restrict__ data,
                      const float* __restrict__ W1d, const float* __restrict__ b1d,
                      const float* __restrict__ W1p, const float* __restrict__ b1p,
                      const float* __restrict__ W1c, const float* __restrict__ b1c,
                      const float* __restrict__ W2d, const float* __restrict__ b2d,
                      const float* __restrict__ W2c, const float* __restrict__ b2c,
                      const float* __restrict__ pb,
                      float* __restrict__ out) {
  __shared__ _Float16 A1[32 * K1];        // [b][ ctx(180) | x_t(6) | pad ]  12 KB
  __shared__ _Float16 A2[32 * K2];        // [b][ h(240) | pad ]             16 KB
  __shared__ _Float16 xbuf[C_DIM * XCH];  // staged input chunk            22.5 KB

  const int tid  = threadIdx.x;
  const int wave = tid >> 5;
  const int lane = tid & 31;

  // ---- zero the LDS A buffers (pads must stay 0 forever) ----
  for (int i = tid; i < 32 * K1; i += NTHREADS) A1[i] = (_Float16)0.0f;
  for (int i = tid; i < 32 * K2; i += NTHREADS) A2[i] = (_Float16)0.0f;

  // ---- build B-fragments in VGPRs (held live across the whole T loop) ----
  // B-fragment layout used here (32x16 f16, k order as in load_a_frag): lane L(<16): N=L, K=0..15
  // contiguous; lane L+16: N=L, K=16..31.  W1c[h][c] / W2c[c][h] are already
  // [n][k] row-major so the gather is along contiguous k.
  v16h b1frag[6];
  v8f  cinit1[2] = {v8f{}, v8f{}};
  if (wave < 15) {
    const int n = wave * 16 + (lane & 15);           // h column, 0..239
#pragma unroll
    for (int kc = 0; kc < 6; ++kc) {
      const int kb = kc * 32 + (lane >> 4) * 16;
      v16h f;
#pragma unroll
      for (int i = 0; i < 16; ++i) {
        const int k = kb + i;
        float v = 0.0f;
        if (k < C_DIM)            v = W1c[n * C_DIM + k];          // ctx weights
        else if (k < C_DIM + D_DIM) v = W1d[n * D_DIM + (k - C_DIM)]; // x weights
        f[i] = (_Float16)v;
      }
      b1frag[kc] = f;
    }
    // Accumulator init = b1d + b1p + b1c + pb @ W1p.T  (per (b,h) in C layout)
#pragma unroll
    for (int mt = 0; mt < 2; ++mt) {
      v8f c = v8f{};
#pragma unroll
      for (int v = 0; v < 8; ++v) {
        const int b = mt * 16 + v + (lane >> 4) * 8;
        float val = 0.0f;
        if (b < B_N)
          val = b1d[n] + b1p[n] + b1c[n]
              + pb[b * 2 + 0] * W1p[n * 2 + 0]
              + pb[b * 2 + 1] * W1p[n * 2 + 1];
        c[v] = val;
      }
      cinit1[mt] = c;
    }
  }

  v16h b2frag[8];
  v8f  cinit2 = v8f{};
  if (wave < 12) {
    const int n = wave * 16 + (lane & 15);           // 0..179: ctx, 180..185: out
#pragma unroll
    for (int kc = 0; kc < 8; ++kc) {
      const int kb = kc * 32 + (lane >> 4) * 16;
      v16h f;
#pragma unroll
      for (int i = 0; i < 16; ++i) {
        const int k = kb + i;
        float v = 0.0f;
        if (k < H_DIM) {
          if (n < C_DIM)            v = W2c[n * H_DIM + k];
          else if (n < C_DIM + D_DIM) v = W2d[(n - C_DIM) * H_DIM + k];
        }
        f[i] = (_Float16)v;
      }
      b2frag[kc] = f;
    }
    float bias = 0.0f;
    if (n < C_DIM)              bias = b2c[n];
    else if (n < C_DIM + D_DIM) bias = b2d[n - C_DIM];
#pragma unroll
    for (int v = 0; v < 8; ++v) cinit2[v] = bias;
  }
  __syncthreads();

  // ================= sequential recurrence =================
  for (int t = 0; t < T_LEN; ++t) {
    // ---- refill staged input chunk every XCH steps (coalesced along t) ----
    if ((t & (XCH - 1)) == 0) {
      __syncthreads();
      for (int i = tid; i < C_DIM * XCH; i += NTHREADS) {
        const int row = i >> 6;            // b*6+d
        const int tt  = i & (XCH - 1);
        xbuf[i] = (_Float16)data[row * T_LEN + t + tt];
      }
      __syncthreads();
    }

    // ---- scatter x_t into the K-extension columns of A1 ----
    if (tid < B_N * D_DIM) {
      const int b = tid / D_DIM, d = tid - b * D_DIM;
      A1[b * K1 + C_DIM + d] = xbuf[tid * XCH + (t & (XCH - 1))];
    }
    __syncthreads();   // A1 (ctx from prev step + x_t) complete; A2 free

    // ---- GEMM1: h = relu([ctx|x] @ [W1c|W1d]^T + bias1) ----
    if (wave < 15) {
      v8f acc0 = cinit1[0], acc1 = cinit1[1];
#pragma unroll
      for (int kc = 0; kc < 6; ++kc) {
        v16h a0 = load_a_frag(A1, K1, 0,  kc, lane);
        v16h a1 = load_a_frag(A1, K1, 16, kc, lane);
        acc0 = wmma16(a0, b1frag[kc], acc0);
        acc1 = wmma16(a1, b1frag[kc], acc1);
      }
      const int col   = wave * 16 + (lane & 15);
      const int rbase = (lane >> 4) * 8;
#pragma unroll
      for (int v = 0; v < 8; ++v) {
        const float h0 = acc0[v] > 0.0f ? acc0[v] : 0.0f;
        const float h1 = acc1[v] > 0.0f ? acc1[v] : 0.0f;
        A2[(rbase + v)      * K2 + col] = (_Float16)h0;
        A2[(16 + rbase + v) * K2 + col] = (_Float16)h1;
      }
    }
    __syncthreads();   // A2 (h) complete; A1 ctx columns free to overwrite

    // ---- GEMM2: [new_ctx | out_t] = h @ [W2c ; W2d]^T + bias2 ----
    if (wave < 12) {
      v8f acc0 = cinit2, acc1 = cinit2;
#pragma unroll
      for (int kc = 0; kc < 8; ++kc) {
        v16h a0 = load_a_frag(A2, K2, 0,  kc, lane);
        v16h a1 = load_a_frag(A2, K2, 16, kc, lane);
        acc0 = wmma16(a0, b2frag[kc], acc0);
        acc1 = wmma16(a1, b2frag[kc], acc1);
      }
      const int ncol  = wave * 16 + (lane & 15);
      const int rbase = (lane >> 4) * 8;
      if (ncol < C_DIM) {                       // new ctx -> A1 for next step
#pragma unroll
        for (int v = 0; v < 8; ++v) {
          A1[(rbase + v)      * K1 + ncol] = (_Float16)acc0[v];
          A1[(16 + rbase + v) * K1 + ncol] = (_Float16)acc1[v];
        }
      } else if (ncol < C_DIM + D_DIM) {        // out_t -> global
        const int j = ncol - C_DIM;
#pragma unroll
        for (int v = 0; v < 8; ++v) {
          const int b0 = rbase + v, b1i = 16 + rbase + v;
          if (b0  < B_N) out[(b0  * T_LEN + t) * D_DIM + j] = acc0[v];
          if (b1i < B_N) out[(b1i * T_LEN + t) * D_DIM + j] = acc1[v];
        }
      }
    }
    // next iteration's post-x-copy barrier orders A1 writes vs GEMM1 reads
  }
}

extern "C" void kernel_launch(void* const* d_in, const int* in_sizes, int n_in,
                              void* d_out, int out_size, void* d_ws, size_t ws_size,
                              hipStream_t stream) {
  (void)in_sizes; (void)n_in; (void)d_ws; (void)ws_size; (void)out_size;
  const float* data = (const float*)d_in[0];
  const float* W1d  = (const float*)d_in[1];
  const float* b1d  = (const float*)d_in[2];
  const float* W1p  = (const float*)d_in[3];
  const float* b1p  = (const float*)d_in[4];
  const float* W1c  = (const float*)d_in[5];
  const float* b1c  = (const float*)d_in[6];
  const float* W2d  = (const float*)d_in[7];
  const float* b2d  = (const float*)d_in[8];
  const float* W2c  = (const float*)d_in[9];
  const float* b2c  = (const float*)d_in[10];
  const float* pb   = (const float*)d_in[11];
  float* out = (float*)d_out;

  // Single persistent workgroup: the recurrence is a serial chain; one WGP,
  // 16 wave32 waves (4 per SIMD32), weights resident in VGPRs.
  rnnpb_persistent<<<dim3(1), dim3(NTHREADS), 0, stream>>>(
      data, W1d, b1d, W1p, b1p, W1c, b1c, W2d, b2d, W2c, b2c, pb, out);
}
